// GATLayer_48576080118036
// MI455X (gfx1250) — matrix operand order repaired and checked by reading, not compiled, not run on hardware
//
#include <hip/hip_runtime.h>
#include <hip/hip_bf16.h>

// GAT layer for MI455X (gfx1250), wave32.
// Pipeline:
//   Kc: one-shot f32->bf16 conversion of h and W (streaming, off critical path)
//   K1: z = hb @ Wb^T via v_wmma_f32_16x16x32_bf16 -- pure b128 loads + wmma
//   K2: per-node scores s_src / s_dst (wave-per-node dot products)
//   K0: init (zero h_out, encoded -inf for segment max, zero denom)
//   K3: edge score + leaky-relu + segment max (encoded u32 atomicMax)
//   K4: exp(e - m[dst]) + segment sum (f32 atomicAdd)
//   K5: h_out[dst] += alpha * z[src]  (wave-per-edge, float4 gather, f32 atomics)

#define D 128            // D_IN == D_OUT
#define NEG_SLOPE 0.01f

typedef __attribute__((ext_vector_type(16))) __bf16 v16bf;
typedef __attribute__((ext_vector_type(8)))  float  v8f;

__device__ __forceinline__ unsigned short f2bf(float f) {
    unsigned u = __float_as_uint(f);
    unsigned r = u + 0x7fffu + ((u >> 16) & 1u);   // round-to-nearest-even
    return (unsigned short)(r >> 16);
}

// monotone order-preserving float->uint encoding (for atomicMax over signed floats)
__device__ __forceinline__ unsigned enc_f32(float f) {
    unsigned u = __float_as_uint(f);
    return (u & 0x80000000u) ? ~u : (u | 0x80000000u);
}
__device__ __forceinline__ float dec_f32(unsigned e) {
    unsigned u = (e & 0x80000000u) ? (e ^ 0x80000000u) : ~e;
    return __uint_as_float(u);
}
#define ENC_NEG_INF 0x007FFFFFu   // enc_f32(-inf)

// ---------------- Kc: streaming f32 -> bf16 (8 elements / thread) ----------------
__global__ void kc_f32_to_bf16(const float* __restrict__ src,
                               unsigned short* __restrict__ dst, long n8) {
    long i = (long)blockIdx.x * blockDim.x + threadIdx.x;
    if (i >= n8) return;
    const float4* s = (const float4*)src + 2 * i;
    float4 a = s[0], b = s[1];
    union { unsigned short u[8]; uint4 q; } o;
    o.u[0] = f2bf(a.x); o.u[1] = f2bf(a.y); o.u[2] = f2bf(a.z); o.u[3] = f2bf(a.w);
    o.u[4] = f2bf(b.x); o.u[5] = f2bf(b.y); o.u[6] = f2bf(b.z); o.u[7] = f2bf(b.w);
    ((uint4*)dst)[i] = o.q;
}

// ---------------- K1: z = hb @ Wb^T with WMMA ----------------
// grid.x = ceil(N/16) M-tiles; 256 threads = 8 waves; wave w owns N-tile w.
// Fragments are contiguous bf16 runs -> 4x global_load_b128 + 1 wmma per K-step.
__global__ void k1_gemm_wmma(const unsigned short* __restrict__ hb,
                             const unsigned short* __restrict__ Wb,  // [D][D] bf16, row = out dim
                             float* __restrict__ z, int N) {
    const int wave = threadIdx.x >> 5;
    const int lane = threadIdx.x & 31;
    const int half = lane >> 4;
    const int mr   = lane & 15;
    const int m0   = blockIdx.x * 16;
    const int n0   = wave * 16;

    int arow = m0 + mr; if (arow >= N) arow = N - 1;     // duplicate-read only on partial tile
    const unsigned short* ap = hb + (size_t)arow * D;
    const unsigned short* bp = Wb + (size_t)(n0 + mr) * D;

    v8f acc = {};
    union { v16bf v; uint4 q[2]; } A, B;

    #pragma unroll
    for (int kb = 0; kb < D; kb += 32) {
        // A 16x32 bf16: VGPR0-7 = K [16h, 16h+15]
        A.q[0] = *(const uint4*)(ap + kb + 16 * half);
        A.q[1] = *(const uint4*)(ap + kb + 16 * half + 8);
        // B 32x16 bf16 (B[k][n] = W[n][k]): VGPR0-7 = K [16h, 16h+15]
        B.q[0] = *(const uint4*)(bp + kb + 16 * half);
        B.q[1] = *(const uint4*)(bp + kb + 16 * half + 8);
        acc = __builtin_amdgcn_wmma_f32_16x16x32_bf16(
            false, A.v, false, B.v, (short)0, acc, false, false);
    }

    // C/D layout: VGPR r -> M = r + 8*half, N = mr.
    // Fast path (full tile, the only case for N%16==0): one base pointer,
    // 8 stores at immediate offsets r*D*4 = r*512 bytes.
    float* zp = z + (size_t)(m0 + 8 * half) * D + n0 + mr;
    if (m0 + 16 <= N) {
        #pragma unroll
        for (int r = 0; r < 8; ++r) zp[(size_t)r * D] = acc[r];
    } else {
        #pragma unroll
        for (int r = 0; r < 8; ++r)
            if (m0 + r + 8 * half < N) zp[(size_t)r * D] = acc[r];
    }
}

// ---------------- K2: per-node scalar scores ----------------
__global__ void k2_node_scores(const float* __restrict__ z,
                               const float* __restrict__ emb,
                               const float* __restrict__ wfc,
                               const float* __restrict__ wemb,
                               float* __restrict__ s_src,
                               float* __restrict__ s_dst, int N) {
    int v = blockIdx.x * 8 + (threadIdx.x >> 5);
    if (v >= N) return;
    int lane = threadIdx.x & 31;
    float fs = 0.f, fd = 0.f;
    #pragma unroll
    for (int k = lane; k < D; k += 32) {
        float zv = z[(size_t)v * D + k];
        float ev = emb[(size_t)v * D + k];
        fs = fmaf(zv, wfc[k],     fmaf(ev, wemb[k],     fs));
        fd = fmaf(zv, wfc[D + k], fmaf(ev, wemb[D + k], fd));
    }
    #pragma unroll
    for (int off = 16; off > 0; off >>= 1) {
        fs += __shfl_xor(fs, off, 32);
        fd += __shfl_xor(fd, off, 32);
    }
    if (lane == 0) { s_src[v] = fs; s_dst[v] = fd; }
}

// ---------------- K0: init ----------------
__global__ void k0_init(float* __restrict__ h_out, unsigned* __restrict__ m_enc,
                        float* __restrict__ denom, int N) {
    size_t idx = (size_t)blockIdx.x * blockDim.x + threadIdx.x;
    if (idx < (size_t)N * D) h_out[idx] = 0.f;
    if (idx < (size_t)N) { m_enc[idx] = ENC_NEG_INF; denom[idx] = 0.f; }
}

// ---------------- K3: edge score + segment max ----------------
__global__ void k3_edge_max(const float* __restrict__ s_src,
                            const float* __restrict__ s_dst,
                            const int* __restrict__ src,
                            const int* __restrict__ dst,
                            float* __restrict__ e_val,
                            unsigned* __restrict__ m_enc, int E) {
    int e = blockIdx.x * blockDim.x + threadIdx.x;
    if (e >= E) return;
    float a = s_src[src[e]] + s_dst[dst[e]];
    float lr = (a > 0.f) ? a : NEG_SLOPE * a;
    e_val[e] = lr;
    atomicMax(&m_enc[dst[e]], enc_f32(lr));
}

// ---------------- K4: exp + segment sum ----------------
__global__ void k4_edge_exp(const float* __restrict__ e_val,
                            const int* __restrict__ dst,
                            const unsigned* __restrict__ m_enc,
                            float* __restrict__ e_exp,
                            float* __restrict__ denom, int E) {
    int e = blockIdx.x * blockDim.x + threadIdx.x;
    if (e >= E) return;
    int d = dst[e];
    float ex = __expf(e_val[e] - dec_f32(m_enc[d]));
    e_exp[e] = ex;
    atomicAdd(&denom[d], ex);
}

// ---------------- K5: weighted aggregation ----------------
// one wave per edge; lane covers 4 contiguous dims via float4; z/h_out are L2-resident
__global__ void k5_aggregate(const float* __restrict__ z,
                             const int* __restrict__ src,
                             const int* __restrict__ dst,
                             const float* __restrict__ e_exp,
                             const float* __restrict__ denom,
                             float* __restrict__ h_out, int E) {
    int e = blockIdx.x * 8 + (threadIdx.x >> 5);
    if (e >= E) return;
    int lane = threadIdx.x & 31;
    int s = src[e], d = dst[e];
    float alpha = e_exp[e] / denom[d];
    float4 zv = *(const float4*)(z + (size_t)s * D + lane * 4);
    float* o = h_out + (size_t)d * D + lane * 4;
    atomicAdd(o + 0, alpha * zv.x);
    atomicAdd(o + 1, alpha * zv.y);
    atomicAdd(o + 2, alpha * zv.z);
    atomicAdd(o + 3, alpha * zv.w);
}

extern "C" void kernel_launch(void* const* d_in, const int* in_sizes, int n_in,
                              void* d_out, int out_size, void* d_ws, size_t ws_size,
                              hipStream_t stream) {
    const float* h    = (const float*)d_in[0];
    const float* emb  = (const float*)d_in[1];
    const float* W    = (const float*)d_in[2];
    const float* wfc  = (const float*)d_in[3];
    const float* wemb = (const float*)d_in[4];
    const int*   src  = (const int*)d_in[5];
    const int*   dst  = (const int*)d_in[6];
    float* h_out = (float*)d_out;

    const int N = in_sizes[0] / D;
    const int E = in_sizes[5];

    // workspace layout (16B-aligned pieces)
    char* ws = (char*)d_ws;
    float*          z     = (float*)ws;          ws += (size_t)N * D * sizeof(float);
    unsigned short* hb    = (unsigned short*)ws; ws += (size_t)N * D * sizeof(unsigned short);
    float*          e_val = (float*)ws;          ws += (size_t)E * sizeof(float);
    float*          e_exp = (float*)ws;          ws += (size_t)E * sizeof(float);
    unsigned short* Wb    = (unsigned short*)ws; ws += (size_t)D * D * sizeof(unsigned short);
    float*          s_src = (float*)ws;          ws += (size_t)N * sizeof(float);
    float*          s_dst = (float*)ws;          ws += (size_t)N * sizeof(float);
    unsigned*       m_enc = (unsigned*)ws;       ws += (size_t)N * sizeof(unsigned);
    float*          denom = (float*)ws;

    // Kc: convert h and W to bf16 (one streaming pass each)
    {
        long n8 = (long)N * D / 8;
        kc_f32_to_bf16<<<(unsigned)((n8 + 255) / 256), 256, 0, stream>>>(h, hb, n8);
        long w8 = (long)D * D / 8;
        kc_f32_to_bf16<<<(unsigned)((w8 + 255) / 256), 256, 0, stream>>>(W, Wb, w8);
    }
    // K0: init h_out / m / denom
    {
        size_t total = (size_t)N * D;
        k0_init<<<(unsigned)((total + 255) / 256), 256, 0, stream>>>(h_out, m_enc, denom, N);
    }
    // K1: z = hb @ Wb^T (WMMA, pure b128 loads)
    k1_gemm_wmma<<<(N + 15) / 16, 256, 0, stream>>>(hb, Wb, z, N);
    // K2: node scores
    k2_node_scores<<<(N + 7) / 8, 256, 0, stream>>>(z, emb, wfc, wemb, s_src, s_dst, N);
    // K3: edge scores + segment max
    k3_edge_max<<<(E + 255) / 256, 256, 0, stream>>>(s_src, s_dst, src, dst, e_val, m_enc, E);
    // K4: exp + segment sum
    k4_edge_exp<<<(E + 255) / 256, 256, 0, stream>>>(e_val, dst, m_enc, e_exp, denom, E);
    // K5: weighted neighbor aggregation
    k5_aggregate<<<(E + 7) / 8, 256, 0, stream>>>(z, src, dst, e_exp, denom, h_out, E);
}
